// ToySAE_40544491274571
// MI455X (gfx1250) — matrix operand order repaired and checked by reading, not compiled, not run on hardware
//
#include <hip/hip_runtime.h>

#define BATCH 4096
#define NIN   768
#define NHID  16384
#define TOPK  32
#define KPAD  40   // 32 bf16 + 8 pad per LDS row = 80B (16B aligned, conflict-free)

typedef __attribute__((ext_vector_type(16))) __bf16 v16bf;
typedef __attribute__((ext_vector_type(8)))  __bf16 v8bf;
typedef __attribute__((ext_vector_type(8)))  float  v8f;

typedef int v4i __attribute__((vector_size(16)));
typedef __attribute__((address_space(1))) v4i g_v4i;   // global 16B chunk
typedef __attribute__((address_space(3))) v4i l_v4i;   // LDS 16B chunk

__device__ __forceinline__ __bf16 tobf(float f) { return (__bf16)f; }

__device__ __forceinline__ v16bf pack16(v8bf lo, v8bf hi) {
  v16bf r;
#pragma unroll
  for (int i = 0; i < 8; i++) { r[i] = lo[i]; r[8 + i] = hi[i]; }
  return r;
}

// ---------------------------------------------------------------------------
// Kernel 0a: x (f32) -> xb (bf16), same layout [B, NIN]
// ---------------------------------------------------------------------------
__global__ __launch_bounds__(256) void cvt_x_bf16(const float* __restrict__ x,
                                                  __bf16* __restrict__ xb)
{
  const size_t i = ((size_t)blockIdx.x * 256 + threadIdx.x) * 8;
  const float4 f0 = *(const float4*)(x + i);
  const float4 f1 = *(const float4*)(x + i + 4);
  v8bf h;
  h[0]=tobf(f0.x); h[1]=tobf(f0.y); h[2]=tobf(f0.z); h[3]=tobf(f0.w);
  h[4]=tobf(f1.x); h[5]=tobf(f1.y); h[6]=tobf(f1.z); h[7]=tobf(f1.w);
  *(v8bf*)(xb + i) = h;
}

// ---------------------------------------------------------------------------
// Kernel 0b: W_enc [NIN, NHID] f32 -> WT [NHID, NIN] bf16 (tiled transpose)
// ---------------------------------------------------------------------------
__global__ __launch_bounds__(256) void cvt_wenc_t_bf16(const float* __restrict__ W,
                                                       __bf16* __restrict__ WT)
{
  __shared__ float tile[32][33];
  const int t  = threadIdx.x;
  const int tx = t & 31, ty = t >> 5;           // 32 x 8
  const int n0 = blockIdx.x * 32;               // NHID dim
  const int k0 = blockIdx.y * 32;               // NIN dim
#pragma unroll
  for (int i = 0; i < 4; i++)                   // coalesced read over n
    tile[ty + i * 8][tx] = W[(size_t)(k0 + ty + i * 8) * NHID + n0 + tx];
  __syncthreads();
#pragma unroll
  for (int i = 0; i < 4; i++)                   // coalesced write over k
    WT[(size_t)(n0 + ty + i * 8) * NIN + k0 + tx] = tobf(tile[tx][ty + i * 8]);
}

// ---------------------------------------------------------------------------
// Kernel 1 (fast path): preact = x @ W_enc + b_enc from pre-converted bf16,
// async global->LDS copies (ASYNCcnt) with double buffering + bf16 WMMA.
// Block tile 128x128, 8 waves, each wave 2x4 of 16x16 tiles.
// ---------------------------------------------------------------------------
__global__ __launch_bounds__(256) void sae_encode_wmma_async(
    const __bf16* __restrict__ xb, const __bf16* __restrict__ WT,
    const float* __restrict__ benc, float* __restrict__ preact)
{
  __shared__ __align__(16) __bf16 Alds[2][128 * KPAD];  // [m][k]
  __shared__ __align__(16) __bf16 Blds[2][128 * KPAD];  // [n][k]

  const int t    = threadIdx.x;
  const int lane = t & 31;
  const int wave = t >> 5;
  const int l16  = lane & 15;
  const int hs   = lane >> 4;
  const int wrow = wave >> 1;
  const int wcol = wave & 1;
  const int row0 = blockIdx.y * 128;
  const int col0 = blockIdx.x * 128;

  v8f acc[2][4];
#pragma unroll
  for (int a = 0; a < 2; a++)
#pragma unroll
    for (int b = 0; b < 4; b++)
#pragma unroll
      for (int r = 0; r < 8; r++) acc[a][b][r] = 0.0f;

  // Stage one 128x32 bf16 tile pair via async 16B chunks: 4 per thread.
  auto stage = [&](int buf, int kk) {
#pragma unroll
    for (int c = 0; c < 2; c++) {
      const int id  = c * 256 + t;
      const int row = id >> 2;        // 0..127
      const int cc  = id & 3;         // 16B chunk within 64B row
      const __bf16* ga = xb + (size_t)(row0 + row) * NIN + kk + cc * 8;
      __bf16*       la = &Alds[buf][row * KPAD + cc * 8];
      __builtin_amdgcn_global_load_async_to_lds_b128((g_v4i*)ga, (l_v4i*)la, 0, 0);
      const __bf16* gb = WT + (size_t)(col0 + row) * NIN + kk + cc * 8;
      __bf16*       lb = &Blds[buf][row * KPAD + cc * 8];
      __builtin_amdgcn_global_load_async_to_lds_b128((g_v4i*)gb, (l_v4i*)lb, 0, 0);
    }
  };

  stage(0, 0);
  int cur = 0;
  for (int kk = 0; kk < NIN; kk += 32) {
    __builtin_amdgcn_s_wait_asynccnt(0);    // my chunks for buf[cur] landed
    __syncthreads();                        // everyone's chunks landed
    if (kk + 32 < NIN) stage(cur ^ 1, kk + 32);  // overlap copy with compute

    v16bf afrag[2], bfrag[4];
#pragma unroll
    for (int mt = 0; mt < 2; mt++) {        // A frag: 16 contiguous K per half
      const __bf16* ap = &Alds[cur][(wrow * 32 + mt * 16 + l16) * KPAD];
      v8bf lo = *(const v8bf*)(ap + hs * 16);
      v8bf hi = *(const v8bf*)(ap + hs * 16 + 8);
      afrag[mt] = pack16(lo, hi);
    }
#pragma unroll
    for (int nt = 0; nt < 4; nt++) {        // B frag: 16 contiguous K per half
      const __bf16* bp = &Blds[cur][(wcol * 64 + nt * 16 + l16) * KPAD + hs * 16];
      v8bf lo = *(const v8bf*)(bp);
      v8bf hi = *(const v8bf*)(bp + 8);
      bfrag[nt] = pack16(lo, hi);
    }
#pragma unroll
    for (int mt = 0; mt < 2; mt++)
#pragma unroll
      for (int nt = 0; nt < 4; nt++)
        acc[mt][nt] = __builtin_amdgcn_wmma_f32_16x16x32_bf16(
            false, afrag[mt], false, bfrag[nt],
            (short)0, acc[mt][nt], false, false);
    cur ^= 1;
  }

#pragma unroll
  for (int nt = 0; nt < 4; nt++) {
    const int col  = col0 + wcol * 64 + nt * 16 + l16;
    const float bias = benc[col];
#pragma unroll
    for (int mt = 0; mt < 2; mt++) {
      const int rbase = row0 + wrow * 32 + mt * 16 + hs * 8;
#pragma unroll
      for (int r = 0; r < 8; r++)
        preact[(size_t)(rbase + r) * NHID + col] = acc[mt][nt][r] + bias;
    }
  }
}

// ---------------------------------------------------------------------------
// Kernel 1 (fallback, no workspace): fused f32->bf16 conversion in-kernel.
// ---------------------------------------------------------------------------
__global__ __launch_bounds__(256) void sae_encode_wmma(
    const float* __restrict__ x, const float* __restrict__ Wenc,
    const float* __restrict__ benc, float* __restrict__ preact)
{
  __shared__ __align__(16) __bf16 Alds[128 * KPAD];
  __shared__ __align__(16) __bf16 Blds[128 * KPAD];

  const int t    = threadIdx.x;
  const int lane = t & 31;
  const int wave = t >> 5;
  const int l16  = lane & 15;
  const int hs   = lane >> 4;
  const int wrow = wave >> 1;
  const int wcol = wave & 1;
  const int row0 = blockIdx.y * 128;
  const int col0 = blockIdx.x * 128;

  v8f acc[2][4];
#pragma unroll
  for (int a = 0; a < 2; a++)
#pragma unroll
    for (int b = 0; b < 4; b++)
#pragma unroll
      for (int r = 0; r < 8; r++) acc[a][b][r] = 0.0f;

  const int ar   = t >> 1;
  const int aseg = (t & 1) * 16;
  const int bk   = t >> 3;
  const int bseg = (t & 7) * 16;

  const float* aSrc = x    + (size_t)(row0 + ar) * NIN  + aseg;
  const float* bSrc = Wenc + (size_t)bk          * NHID + col0 + bseg;

  for (int kk = 0; kk < NIN; kk += 32) {
    if (kk + 32 < NIN) {
      __builtin_prefetch(aSrc + kk + 32, 0, 1);
      __builtin_prefetch(bSrc + (size_t)(kk + 32) * NHID, 0, 1);
    }
    __syncthreads();
    {
      const float4* s = (const float4*)(aSrc + kk);
      float4 f0 = s[0], f1 = s[1], f2 = s[2], f3 = s[3];
      v8bf h0, h1;
      h0[0]=tobf(f0.x); h0[1]=tobf(f0.y); h0[2]=tobf(f0.z); h0[3]=tobf(f0.w);
      h0[4]=tobf(f1.x); h0[5]=tobf(f1.y); h0[6]=tobf(f1.z); h0[7]=tobf(f1.w);
      h1[0]=tobf(f2.x); h1[1]=tobf(f2.y); h1[2]=tobf(f2.z); h1[3]=tobf(f2.w);
      h1[4]=tobf(f3.x); h1[5]=tobf(f3.y); h1[6]=tobf(f3.z); h1[7]=tobf(f3.w);
      __bf16* dst = &Alds[ar * KPAD + aseg];
      *(v8bf*)(dst)     = h0;
      *(v8bf*)(dst + 8) = h1;
    }
    {
      const float4* s = (const float4*)(bSrc + (size_t)kk * NHID);
      float4 f0 = s[0], f1 = s[1], f2 = s[2], f3 = s[3];
      float v[16] = { f0.x,f0.y,f0.z,f0.w, f1.x,f1.y,f1.z,f1.w,
                      f2.x,f2.y,f2.z,f2.w, f3.x,f3.y,f3.z,f3.w };
#pragma unroll
      for (int j = 0; j < 16; j++)
        Blds[(bseg + j) * KPAD + bk] = tobf(v[j]);
    }
    __syncthreads();

    v16bf afrag[2], bfrag[4];
#pragma unroll
    for (int mt = 0; mt < 2; mt++) {
      const __bf16* ap = &Alds[(wrow * 32 + mt * 16 + l16) * KPAD];
      v8bf lo = *(const v8bf*)(ap + hs * 16);
      v8bf hi = *(const v8bf*)(ap + hs * 16 + 8);
      afrag[mt] = pack16(lo, hi);
    }
#pragma unroll
    for (int nt = 0; nt < 4; nt++) {
      const __bf16* bp = &Blds[(wcol * 64 + nt * 16 + l16) * KPAD + hs * 16];
      v8bf lo = *(const v8bf*)(bp);
      v8bf hi = *(const v8bf*)(bp + 8);
      bfrag[nt] = pack16(lo, hi);
    }
#pragma unroll
    for (int mt = 0; mt < 2; mt++)
#pragma unroll
      for (int nt = 0; nt < 4; nt++)
        acc[mt][nt] = __builtin_amdgcn_wmma_f32_16x16x32_bf16(
            false, afrag[mt], false, bfrag[nt],
            (short)0, acc[mt][nt], false, false);
  }

#pragma unroll
  for (int nt = 0; nt < 4; nt++) {
    const int col  = col0 + wcol * 64 + nt * 16 + l16;
    const float bias = benc[col];
#pragma unroll
    for (int mt = 0; mt < 2; mt++) {
      const int rbase = row0 + wrow * 32 + mt * 16 + hs * 8;
#pragma unroll
      for (int r = 0; r < 8; r++)
        preact[(size_t)(rbase + r) * NHID + col] = acc[mt][nt][r] + bias;
    }
  }
}

// ---------------------------------------------------------------------------
// Kernel 2: per-row TopK(32) over relu(preact). One block (8 waves) per row.
// ---------------------------------------------------------------------------
__global__ __launch_bounds__(256) void sae_topk(
    const float* __restrict__ preact, float* __restrict__ hidden,
    float* __restrict__ wsv, int* __restrict__ wsi)
{
  const int b    = blockIdx.x;
  const int t    = threadIdx.x;
  const int lane = t & 31;
  const int wave = t >> 5;

  float regs[64];
  const float* row = preact + (size_t)b * NHID;
#pragma unroll
  for (int i = 0; i < 64; i++) {
    float v = row[i * 256 + t];
    regs[i] = v > 0.0f ? v : 0.0f;   // relu -> all candidates >= 0
  }

  __shared__ float topv[TOPK];
  __shared__ int   topi[TOPK];
  __shared__ float wv[8];
  __shared__ int   wi[8];
  __shared__ int   winI;

  for (int k = 0; k < TOPK; k++) {
    float bv = -2.0f; int bi = 0;
#pragma unroll
    for (int i = 0; i < 64; i++) {
      float v = regs[i];
      if (v > bv) { bv = v; bi = i * 256 + t; }
    }
#pragma unroll
    for (int off = 16; off > 0; off >>= 1) {
      float ov = __shfl_xor(bv, off, 32);
      int   oi = __shfl_xor(bi, off, 32);
      if (ov > bv) { bv = ov; bi = oi; }
    }
    if (lane == 0) { wv[wave] = bv; wi[wave] = bi; }
    __syncthreads();
    if (t == 0) {
      float best = wv[0]; int besti = wi[0];
      for (int w = 1; w < 8; w++)
        if (wv[w] > best) { best = wv[w]; besti = wi[w]; }
      topv[k] = best; topi[k] = besti; winI = besti;
    }
    __syncthreads();
    const int j = winI;
    if ((j & 255) == t) regs[j >> 8] = -1.0f;   // zap winner (marker < 0)
  }
  __syncthreads();

  float* hrow = hidden + (size_t)b * NHID;
#pragma unroll 4
  for (int i = 0; i < 64; i++) {
    float outv = 0.0f;
    if (regs[i] < 0.0f) {
      const int j = i * 256 + t;
      for (int k = 0; k < TOPK; k++)
        if (topi[k] == j) outv = topv[k];
    }
    __builtin_nontemporal_store(outv, hrow + i * 256 + t);
  }
  if (t < TOPK) {
    wsv[(size_t)b * TOPK + t] = topv[t];
    wsi[(size_t)b * TOPK + t] = topi[t];
  }
}

// ---------------------------------------------------------------------------
// Kernel 3: out[b,:] = relu(b_dec + sum_k val[b,k] * W_dec[idx[b,k], :])
// ---------------------------------------------------------------------------
__global__ __launch_bounds__(256) void sae_decode(
    const float* __restrict__ Wdec, const float* __restrict__ bdec,
    const float* __restrict__ wsv, const int* __restrict__ wsi,
    float* __restrict__ out)
{
  const int b = blockIdx.x;
  const int t = threadIdx.x;
  __shared__ float sv[TOPK];
  __shared__ int   si[TOPK];
  if (t < TOPK) {
    sv[t] = wsv[(size_t)b * TOPK + t];
    si[t] = wsi[(size_t)b * TOPK + t];
  }
  __syncthreads();

  float a0 = bdec[t], a1 = bdec[t + 256], a2 = bdec[t + 512];
#pragma unroll 8
  for (int k = 0; k < TOPK; k++) {
    const float  v = sv[k];
    const size_t r = (size_t)si[k] * NIN;
    a0 += v * Wdec[r + t];
    a1 += v * Wdec[r + t + 256];
    a2 += v * Wdec[r + t + 512];
  }
  float* orow = out + (size_t)b * NIN;
  __builtin_nontemporal_store(fmaxf(a0, 0.0f), orow + t);
  __builtin_nontemporal_store(fmaxf(a1, 0.0f), orow + t + 256);
  __builtin_nontemporal_store(fmaxf(a2, 0.0f), orow + t + 512);
}

// ---------------------------------------------------------------------------
extern "C" void kernel_launch(void* const* d_in, const int* in_sizes, int n_in,
                              void* d_out, int out_size, void* d_ws, size_t ws_size,
                              hipStream_t stream) {
  const float* x    = (const float*)d_in[0];
  const float* Wenc = (const float*)d_in[1];
  const float* benc = (const float*)d_in[2];
  const float* Wdec = (const float*)d_in[3];
  const float* bdec = (const float*)d_in[4];

  float* out    = (float*)d_out;                          // [B, NIN]
  float* hidden = out + (size_t)BATCH * NIN;              // [B, NHID]
  float* preact = hidden + (size_t)BATCH * NHID;          // [B, NHID]

  // workspace layout
  const size_t topkBytes = (size_t)BATCH * TOPK * 4;              // vals
  const size_t xbBytes   = (size_t)BATCH * NIN * 2;               // x bf16
  const size_t wtBytes   = (size_t)NHID * NIN * 2;                // W_enc^T bf16
  float* wsv = (float*)d_ws;
  int*   wsi = (int*)((char*)d_ws + topkBytes);
  __bf16* xb = (__bf16*)((char*)d_ws + 2 * topkBytes);
  __bf16* WT = (__bf16*)((char*)d_ws + 2 * topkBytes + xbBytes);
  const size_t wsNeed = 2 * topkBytes + xbBytes + wtBytes;

  dim3 g1(NHID / 128, BATCH / 128);
  if (ws_size >= wsNeed) {
    cvt_x_bf16<<<(BATCH * NIN) / (256 * 8), 256, 0, stream>>>(x, xb);
    cvt_wenc_t_bf16<<<dim3(NHID / 32, NIN / 32), 256, 0, stream>>>(Wenc, WT);
    sae_encode_wmma_async<<<g1, 256, 0, stream>>>(xb, WT, benc, preact);
  } else {
    sae_encode_wmma<<<g1, 256, 0, stream>>>(x, Wenc, benc, preact);
  }
  sae_topk<<<BATCH, 256, 0, stream>>>(preact, hidden, wsv, wsi);
  sae_decode<<<BATCH, 256, 0, stream>>>(Wdec, bdec, wsv, wsi, out);
}
